// RGAT_PyG_39908836114547
// MI455X (gfx1250) — matrix operand order repaired and checked by reading, not compiled, not run on hardware
//
#include <hip/hip_runtime.h>
#include <hip/hip_bf16.h>

// ---------------- problem constants (match reference) ----------------
#define NN      100000     // nodes
#define NE      200000     // edges per edge type
#define IN_F    1024
#define HF      512
#define NH      4
#define HD      128        // HF / NH
#define NCLS    2983
#define BATCH   1024
#define KC      256        // K-chunk staged in LDS (all GEMM K are multiples)

typedef __attribute__((ext_vector_type(16))) __bf16 v16bf;
typedef __attribute__((ext_vector_type(8)))  __bf16 v8bf;
typedef __attribute__((ext_vector_type(8)))  float  v8f;
typedef __attribute__((ext_vector_type(4)))  unsigned int u32x4;
typedef __attribute__((ext_vector_type(8)))  int i32x8;
typedef __attribute__((ext_vector_type(4)))  int i32x4;

struct bfrag { v8bf lo, hi; };

// ---------------- helpers ----------------
__device__ inline void atomicMaxF(float* addr, float val) {
    unsigned int* ai = (unsigned int*)addr;
    unsigned int old = *ai;
    while (__uint_as_float(old) < val) {
        unsigned int assumed = old;
        old = atomicCAS(ai, assumed, __float_as_uint(val));
        if (old == assumed) break;
    }
}

// ---------------- f32 -> bf16 convert ----------------
__global__ void cvt_f32_bf16_kernel(const float* __restrict__ src,
                                    __bf16* __restrict__ dst, long long n) {
    long long i = (long long)blockIdx.x * blockDim.x + threadIdx.x;
    if (i < n) dst[i] = (__bf16)src[i];
}

// leaky_relu(0.01) then convert to bf16 (between layers)
__global__ void relu_cvt_kernel(const float* __restrict__ src,
                                __bf16* __restrict__ dst, long long n) {
    long long i = (long long)blockIdx.x * blockDim.x + threadIdx.x;
    if (i < n) {
        float v = src[i];
        v = v > 0.0f ? v : 0.01f * v;
        dst[i] = (__bf16)v;
    }
}

// out_accum[n, c] = ba[c] + bb[c]  (HeteroConv sums both convs' biases)
__global__ void bias_init_kernel(float* __restrict__ out,
                                 const float* __restrict__ ba,
                                 const float* __restrict__ bb, long long total) {
    long long i = (long long)blockIdx.x * blockDim.x + threadIdx.x;
    if (i < total) {
        int c = (int)(i & (HF - 1));  // HF = 512
        out[i] = ba[c] + bb[c];
    }
}

// ---------------- weight pack: f32 [K,N] -> bf16 WMMA-fragment order ----------
// Packed layout: ((tile_n * (K/32) + kb) * 32 + lane) * 16 + j
//   where value = B[kb*32 + (lane>>4)*16 + j][tile_n*16 + (lane&15)]
// so each lane's 16 B-fragment values are 32 contiguous bytes.
__global__ void pack_w_kernel(const float* __restrict__ W, __bf16* __restrict__ Bp,
                              int N, int K, int tnp) {
    int t = blockIdx.x * blockDim.x + threadIdx.x;   // (tile, kb, lane)
    int kblocks = K >> 5;
    int total = tnp * kblocks * 32;
    if (t >= total) return;
    int lane = t & 31;
    int kb   = (t >> 5) % kblocks;
    int tile = (t >> 5) / kblocks;
    int col  = tile * 16 + (lane & 15);
    int kbase = kb * 32 + (lane >> 4) * 16;
    __bf16* dst = Bp + (size_t)t * 16;
    if (col < N) {
        const float* srcp = W + (size_t)kbase * N + col;
#pragma unroll
        for (int j = 0; j < 16; ++j) dst[j] = (__bf16)srcp[(size_t)j * N];
    } else {
#pragma unroll
        for (int j = 0; j < 16; ++j) dst[j] = (__bf16)0.0f;
    }
}

// ---------------- WMMA bf16 GEMM, TDM-staged packed B ----------------
// C[M,N] = A[M,K] @ B[K,N] (+bias).  A row-major bf16; B pre-packed (above).
// Block = 8 waves -> 128 rows x 64 cols (4 N-tiles) of C; A fragments are
// reused for 4 WMMAs.  Per 256-wide K chunk, one iterate-mode TDM descriptor
// DMAs the 4 tiles' packed fragments (contiguous runs) into LDS; B fragments
// are then two ds_load_b128 per tile.
__global__ void wmma_gemm_bf16_kernel(const __bf16* __restrict__ A,
                                      const __bf16* __restrict__ Bp,
                                      float* __restrict__ C,
                                      const float* __restrict__ bias, // nullable
                                      int M, int N, int K, int ntg) {
    __shared__ __bf16 Bs[4 * KC * 16];   // 32 KB

    int bm   = blockIdx.x / ntg;
    int tg   = blockIdx.x - bm * ntg;          // group of 4 N tiles
    int lane = threadIdx.x & 31;
    int wv   = threadIdx.x >> 5;
    int half = lane >> 4;
    int kblocks = K >> 5;

    int tile_m = bm * 8 + wv;
    bool active = (tile_m * 16) < M;
    int row = tile_m * 16 + (lane & 15);

    v8f acc[4];
#pragma unroll
    for (int t = 0; t < 4; ++t) {
        int col = tg * 64 + t * 16 + (lane & 15);
        float bv = (bias && col < N) ? bias[col] : 0.0f;
#pragma unroll
        for (int g = 0; g < 8; ++g) acc[t][g] = bv;
    }

    const __bf16* arow = A + (size_t)row * K + half * 16;

    for (int c0 = 0; c0 < K; c0 += KC) {
        __syncthreads();   // previous chunk's LDS reads complete
        int kb0 = c0 >> 5;
        const __bf16* src = Bp + (((size_t)(tg * 4) * kblocks + kb0) << 9); // *512
#if __has_builtin(__builtin_amdgcn_tensor_load_to_lds)
        if (threadIdx.x < 32) {
            unsigned long long gaddr = (unsigned long long)(const void*)src;
            unsigned ldsOff = (unsigned)(unsigned long long)(uintptr_t)&Bs[0];
            // group0: count=1 | lds_addr | global_addr[56:0] | type=2
            u32x4 g0 = { 1u,
                         ldsOff,
                         (unsigned)(gaddr & 0xFFFFFFFFu),
                         (unsigned)((gaddr >> 32) & 0x01FFFFFFu) | 0x80000000u };
            // group1: data_size=2B, iterate_enable; 1-D tile of KC*16 elems
            unsigned chunk = KC * 16;               // 4096 elements
            i32x8 g1 = { (int)(0x00010000u | (1u << 19)),
                         (int)((chunk & 0xFFFFu) << 16),          // tensor_dim0 lo
                         (int)((chunk >> 16) | (1u << 16)),       // dim0 hi | tensor_dim1=1
                         (int)((chunk & 0xFFFFu) << 16),          // tile_dim0
                         (int)1,                                  // tile_dim1 = 1
                         (int)chunk,                              // tensor_dim0_stride
                         0, 0 };
            // group2 (iterate mode): lds_inc = chunk, global_inc = K*16, count=3 (4x)
            unsigned ginc = (unsigned)(kblocks << 9);             // K*16 elements
            i32x4 g2 = { 0, (int)chunk, (int)ginc, (int)(3u << 16) };
            i32x4 g3 = { 0, 0, 0, 0 };
#if defined(__clang_major__) && __clang_major__ >= 23
            i32x8 z8 = { 0, 0, 0, 0, 0, 0, 0, 0 };
            __builtin_amdgcn_tensor_load_to_lds(g0, g1, g2, g3, z8, 0);
#else
            __builtin_amdgcn_tensor_load_to_lds(g0, g1, g2, g3, 0);
#endif
            __builtin_amdgcn_s_wait_tensorcnt(0);
        }
#else
        for (int idx = threadIdx.x; idx < 4 * KC * 16; idx += 256) {
            int t = idx / (KC * 16), o = idx - t * (KC * 16);
            Bs[idx] = src[(size_t)t * (kblocks << 9) + o];
        }
#endif
        __syncthreads();

        if (active) {
            __builtin_prefetch(arow + c0 + KC, 0, 0);   // next A chunk
            for (int k0 = 0; k0 < KC; k0 += 32) {
                // A fragment: two 16B contiguous runs per lane
                const v8bf* ap = (const v8bf*)(arow + c0 + k0);
                bfrag af;
                af.lo = ap[0];
                af.hi = ap[1];                  // +8 bf16 elements
                v16bf a = __builtin_bit_cast(v16bf, af);

                int boff = (((k0 >> 5) << 5) + lane) << 4;   // ((k0/32)*32+lane)*16
#pragma unroll
                for (int t = 0; t < 4; ++t) {
                    const v8bf* bp2 = (const v8bf*)&Bs[t * (KC * 16) + boff];
                    bfrag bf2;
                    bf2.lo = bp2[0];
                    bf2.hi = bp2[1];
                    v16bf b = __builtin_bit_cast(v16bf, bf2);
                    acc[t] = __builtin_amdgcn_wmma_f32_16x16x32_bf16(
                        false, a, false, b, (short)0, acc[t], false, false);
                }
            }
        }
    }

    if (active) {
#pragma unroll
        for (int t = 0; t < 4; ++t) {
            int col = tg * 64 + t * 16 + (lane & 15);
            if (col < N) {
#pragma unroll
                for (int g = 0; g < 8; ++g) {
                    int r = tile_m * 16 + g + 8 * half;
                    C[(size_t)r * N + col] = acc[t][g];
                }
            }
        }
    }
}

// ---------------- attention: alpha_src / alpha_dst per (node, head) ----------------
__global__ void alpha_kernel(const float* __restrict__ h,       // [NN, HF]
                             const float* __restrict__ a_src,   // [NH, HD]
                             const float* __restrict__ a_dst,   // [NH, HD]
                             float* __restrict__ asrc,          // [NN*NH]
                             float* __restrict__ adst) {
    int t = blockIdx.x * blockDim.x + threadIdx.x;   // node*NH + head
    if (t >= NN * NH) return;
    int head = t & (NH - 1);
    int node = t >> 2;
    const float* hp = h + (size_t)node * HF + head * HD;
    const float* wsv = a_src + head * HD;
    const float* wdv = a_dst + head * HD;
    float ss = 0.0f, sd = 0.0f;
#pragma unroll 4
    for (int j = 0; j < HD; ++j) {
        float v = hp[j];
        ss += v * wsv[j];
        sd += v * wdv[j];
    }
    asrc[t] = ss;
    adst[t] = sd;
}

__global__ void seg_init_kernel(float* __restrict__ segmax, float* __restrict__ segsum) {
    int i = blockIdx.x * blockDim.x + threadIdx.x;
    if (i < NN * NH) {
        segmax[i] = -3.0e38f;
        segsum[i] = 0.0f;
    }
}

// e = leaky_relu(asrc[src] + adst[dst], 0.2); segment max over dst
__global__ void edge_e_kernel(const float* __restrict__ asrc, const float* __restrict__ adst,
                              const int* __restrict__ src, const int* __restrict__ dst,
                              float* __restrict__ ebuf, float* __restrict__ segmax) {
    int t = blockIdx.x * blockDim.x + threadIdx.x;   // edge*NH + head
    if (t >= NE * NH) return;
    int head = t & (NH - 1);
    int e = t >> 2;
    float v = asrc[src[e] * NH + head] + adst[dst[e] * NH + head];
    v = v > 0.0f ? v : 0.2f * v;
    ebuf[t] = v;
    atomicMaxF(&segmax[dst[e] * NH + head], v);
}

// ex = exp(e - segmax[dst]); segment sum over dst
__global__ void edge_ex_kernel(float* __restrict__ ebuf,
                               const float* __restrict__ segmax,
                               const int* __restrict__ dst,
                               float* __restrict__ segsum) {
    int t = blockIdx.x * blockDim.x + threadIdx.x;
    if (t >= NE * NH) return;
    int head = t & (NH - 1);
    int e = t >> 2;
    float ex = __expf(ebuf[t] - segmax[dst[e] * NH + head]);
    ebuf[t] = ex;
    atomicAdd(&segsum[dst[e] * NH + head], ex);
}

// out[dst] += (ex/segsum[dst]) * h[src]   (one block per edge, 2 channels/thread)
__global__ void aggregate_kernel(const float* __restrict__ ebuf,
                                 const float* __restrict__ segsum,
                                 const float* __restrict__ h,
                                 const int* __restrict__ src, const int* __restrict__ dst,
                                 float* __restrict__ out) {
    int e = blockIdx.x;
    if (e >= NE) return;
    int s = src[e], d = dst[e];
#pragma unroll
    for (int it = 0; it < 2; ++it) {
        int ch = threadIdx.x + it * 256;
        int head = ch >> 7;  // HD = 128
        float alpha = ebuf[e * NH + head] / segsum[d * NH + head];
        atomicAdd(&out[(size_t)d * HF + ch], alpha * h[(size_t)s * HF + ch]);
    }
}

// ---------------- launcher ----------------
extern "C" void kernel_launch(void* const* d_in, const int* in_sizes, int n_in,
                              void* d_out, int out_size, void* d_ws, size_t ws_size,
                              hipStream_t stream) {
    const float* x    = (const float*)d_in[0];
    const int*  e1s   = (const int*)d_in[1];
    const int*  e1d   = (const int*)d_in[2];
    const int*  e2s   = (const int*)d_in[3];
    const int*  e2d   = (const int*)d_in[4];
    const float* w0a  = (const float*)d_in[6];
    const float* as0a = (const float*)d_in[7];
    const float* ad0a = (const float*)d_in[8];
    const float* b0a  = (const float*)d_in[9];
    const float* w0b  = (const float*)d_in[10];
    const float* as0b = (const float*)d_in[11];
    const float* ad0b = (const float*)d_in[12];
    const float* b0b  = (const float*)d_in[13];
    const float* w1a  = (const float*)d_in[14];
    const float* as1a = (const float*)d_in[15];
    const float* ad1a = (const float*)d_in[16];
    const float* b1a  = (const float*)d_in[17];
    const float* w1b  = (const float*)d_in[18];
    const float* as1b = (const float*)d_in[19];
    const float* ad1b = (const float*)d_in[20];
    const float* b1b  = (const float*)d_in[21];
    const float* linw = (const float*)d_in[22];
    const float* linb = (const float*)d_in[23];

    // ---- workspace carve-up ----
    char* ws = (char*)d_ws;
    size_t off = 0;
    auto carve = [&](size_t bytes) -> char* {
        char* p = ws + off;
        off = (off + bytes + 255) & ~(size_t)255;
        return p;
    };
    __bf16* XB   = (__bf16*)carve((size_t)NN * IN_F * 2);   // x bf16; reused for h1 bf16
    __bf16* WPA  = (__bf16*)carve((size_t)32 * 32 * 512 * 2);   // packed weight A (<=1MB)
    __bf16* WPB  = (__bf16*)carve((size_t)32 * 32 * 512 * 2);   // packed weight B
    __bf16* LWP  = (__bf16*)carve((size_t)188 * 16 * 512 * 2);  // packed classifier W
    float*  HA   = (float*)carve((size_t)NN * HF * 4);      // h for conv a
    float*  HB   = (float*)carve((size_t)NN * HF * 4);      // h for conv b
    float*  ACC  = (float*)carve((size_t)NN * HF * 4);      // summed conv output
    float*  ASRC = (float*)carve((size_t)NN * NH * 4);
    float*  ADST = (float*)carve((size_t)NN * NH * 4);
    float*  SMAX = (float*)carve((size_t)NN * NH * 4);
    float*  SSUM = (float*)carve((size_t)NN * NH * 4);
    float*  EBUF = (float*)carve((size_t)NE * NH * 4);
    __bf16* H2B  = (__bf16*)carve((size_t)BATCH * HF * 2);  // seed rows bf16

    const int T = 256;
    auto blocks_for = [](long long n, int t) { return (unsigned)((n + t - 1) / t); };

    auto pack = [&](const float* W, __bf16* Bp, int N, int K) {
        int tn = (N + 15) / 16, tnp = 4 * ((tn + 3) / 4);
        long long total = (long long)tnp * (K >> 5) * 32;
        pack_w_kernel<<<blocks_for(total, T), T, 0, stream>>>(W, Bp, N, K, tnp);
    };

    // ---- GEMM launcher: block = 8 waves = 128 x 64 C slab ----
    auto gemm = [&](const __bf16* A, const __bf16* Bp, float* C, const float* bias,
                    int M, int N, int K) {
        int tn = (N + 15) / 16, tnp = 4 * ((tn + 3) / 4), ntg = tnp / 4;
        int tm = (M + 15) / 16, mblocks = (tm + 7) / 8;
        wmma_gemm_bf16_kernel<<<(unsigned)(mblocks * ntg), 256, 0, stream>>>(
            A, Bp, C, bias, M, N, K, ntg);
    };

    // ---- one GATConv accumulation pass into ACC ----
    auto conv = [&](const float* h, const float* a_s, const float* a_d,
                    const int* es, const int* ed) {
        alpha_kernel<<<blocks_for((long long)NN * NH, T), T, 0, stream>>>(h, a_s, a_d, ASRC, ADST);
        seg_init_kernel<<<blocks_for((long long)NN * NH, T), T, 0, stream>>>(SMAX, SSUM);
        edge_e_kernel<<<blocks_for((long long)NE * NH, T), T, 0, stream>>>(ASRC, ADST, es, ed, EBUF, SMAX);
        edge_ex_kernel<<<blocks_for((long long)NE * NH, T), T, 0, stream>>>(EBUF, SMAX, ed, SSUM);
        aggregate_kernel<<<NE, 256, 0, stream>>>(EBUF, SSUM, h, es, ed, ACC);
    };

    // ================= layer 0 =================
    cvt_f32_bf16_kernel<<<blocks_for((long long)NN * IN_F, T), T, 0, stream>>>(x, XB, (long long)NN * IN_F);
    pack(w0a, WPA, HF, IN_F);
    pack(w0b, WPB, HF, IN_F);
    gemm(XB, WPA, HA, nullptr, NN, HF, IN_F);
    gemm(XB, WPB, HB, nullptr, NN, HF, IN_F);
    bias_init_kernel<<<blocks_for((long long)NN * HF, T), T, 0, stream>>>(ACC, b0a, b0b, (long long)NN * HF);
    conv(HA, as0a, ad0a, e1s, e1d);
    conv(HB, as0b, ad0b, e2s, e2d);
    // h1 = leaky_relu(ACC, 0.01) -> bf16 (reuse XB)
    relu_cvt_kernel<<<blocks_for((long long)NN * HF, T), T, 0, stream>>>(ACC, XB, (long long)NN * HF);

    // ================= layer 1 =================
    pack(w1a, WPA, HF, HF);
    pack(w1b, WPB, HF, HF);
    gemm(XB, WPA, HA, nullptr, NN, HF, HF);
    gemm(XB, WPB, HB, nullptr, NN, HF, HF);
    bias_init_kernel<<<blocks_for((long long)NN * HF, T), T, 0, stream>>>(ACC, b1a, b1b, (long long)NN * HF);
    conv(HA, as1a, ad1a, e1s, e1d);
    conv(HB, as1b, ad1b, e2s, e2d);

    // ================= classification head =================
    cvt_f32_bf16_kernel<<<blocks_for((long long)BATCH * HF, T), T, 0, stream>>>(ACC, H2B, (long long)BATCH * HF);
    pack(linw, LWP, NCLS, HF);
    gemm(H2B, LWP, (float*)d_out, linb, BATCH, NCLS, HF);
}
